// ResDeformBlock_v1_66709432041961
// MI455X (gfx1250) — compile-verified, matrix operand order checked by reading, not run on hardware
//
#include <hip/hip_runtime.h>

typedef __attribute__((ext_vector_type(16))) _Float16 v16h;
typedef __attribute__((ext_vector_type(8)))  _Float16 v8h;
typedef __attribute__((ext_vector_type(8)))  float    v8f;

#define HH   96
#define WW2  96
#define HW   (HH*WW2)        // 9216
#define CIN  128
#define COUT 128
#define KTAP 9
#define CK   (CIN*KTAP)      // 1152
#define NB   2

#define TILE_P 16
#define KCH    64            // K-chunk per LDS weight stage (2 WMMAs of K=32), 8 granules/row
#define SROW   (CK + 8)      // 1160 halves; sample-tile row stride (16B aligned rows)
#define WROW   (KCH + 8)     // 72 halves; weight-chunk row stride (144B, 16B aligned)

__device__ __forceinline__ int iclamp(int v, int lo, int hi) {
    return v < lo ? lo : (v > hi ? hi : v);
}

// ---------------- weight fp32 -> fp16 cast (layout already GEMM-ready) -----
__global__ __launch_bounds__(256) void wcast_kernel(const float* __restrict__ w1,
                                                    const float* __restrict__ w2,
                                                    _Float16* __restrict__ o1,
                                                    _Float16* __restrict__ o2) {
    int i = blockIdx.x * 256 + threadIdx.x;
    const int n = COUT * CK;               // 147456
    if (i < n)            o1[i]     = (_Float16)w1[i];
    else                  o2[i - n] = (_Float16)w2[i - n];
}

// ---------------- offset conv3x3: CIN -> 18 channels ------------------------
__global__ __launch_bounds__(256) void offconv_kernel(const float* __restrict__ src,
                                                      const float* __restrict__ w,
                                                      const float* __restrict__ bias,
                                                      float* __restrict__ dst) {
    int gid = blockIdx.x * 256 + threadIdx.x;    // B*HW threads total
    int b   = gid / HW;
    int pix = gid - b * HW;
    int py  = pix / WW2;
    int px  = pix - py * WW2;

    float acc[18];
#pragma unroll
    for (int o = 0; o < 18; ++o) acc[o] = bias[o];

    const float* srcb = src + (size_t)b * CIN * HW;
    for (int c = 0; c < CIN; ++c) {
        const float* xc = srcb + (size_t)c * HW;
        float nb[9];
#pragma unroll
        for (int k = 0; k < 9; ++k) {
            int yy = py + k / 3 - 1;
            int xx = px + (k % 3) - 1;
            bool ok = (yy >= 0) && (yy < HH) && (xx >= 0) && (xx < WW2);
            nb[k] = ok ? xc[yy * WW2 + xx] : 0.f;
        }
        const float* wc = w + (size_t)c * 9;     // w[o][c][k]; o-stride = CIN*9
#pragma unroll
        for (int o = 0; o < 18; ++o) {
#pragma unroll
            for (int k = 0; k < 9; ++k)
                acc[o] += nb[k] * wc[(size_t)o * CIN * 9 + k];   // wave-uniform -> s_load
        }
    }
    float* dp = dst + (size_t)b * 18 * HW + pix;
#pragma unroll
    for (int o = 0; o < 18; ++o) dp[(size_t)o * HW] = acc[o];
}

// ---------------- fused deformable conv: sample -> f16 WMMA GEMM ------------
template <bool ADD_RES>
__global__ __launch_bounds__(256) void deform_kernel(const float* __restrict__ src,
                                                     const float* __restrict__ off,
                                                     const _Float16* __restrict__ wh,
                                                     const float* __restrict__ bias,
                                                     const float* __restrict__ res,
                                                     float* __restrict__ dst) {
    __shared__ _Float16 samp[TILE_P * SROW];   // 37120 B
    __shared__ _Float16 wlds[COUT * WROW];     // 18432 B

    const int tile = blockIdx.x;
    const int b    = tile / (HW / TILE_P);
    const int pix0 = (tile - b * (HW / TILE_P)) * TILE_P;
    const int tid  = threadIdx.x;

    // ---------- phase 1: bilinear sampling into LDS (f16) ----------
    {
        int p  = tid & 15;          // pixel within tile (adjacent lanes -> adjacent x)
        int cg = tid >> 4;          // channel group 0..15
        int pix = pix0 + p;
        int py  = pix / WW2;
        int px  = pix - py * WW2;
        const float* srcb = src + (size_t)b * CIN * HW;
        const float* offb = off + (size_t)b * 18 * HW + pix;
#pragma unroll
        for (int k = 0; k < 9; ++k) {
            float dy = offb[(size_t)(2 * k + 0) * HW];
            float dx = offb[(size_t)(2 * k + 1) * HW];
            float fy = (float)(py + k / 3 - 1) + dy;
            float fx = (float)(px + (k % 3) - 1) + dx;
            float y0f = floorf(fy), x0f = floorf(fx);
            float wy = fy - y0f,   wx = fx - x0f;
            int iy0 = (int)y0f, ix0 = (int)x0f;
            int iy1 = iy0 + 1,  ix1 = ix0 + 1;
            float vy0 = (iy0 >= 0 && iy0 < HH)  ? 1.f : 0.f;
            float vy1 = (iy1 >= 0 && iy1 < HH)  ? 1.f : 0.f;
            float vx0 = (ix0 >= 0 && ix0 < WW2) ? 1.f : 0.f;
            float vx1 = (ix1 >= 0 && ix1 < WW2) ? 1.f : 0.f;
            int cy0 = iclamp(iy0, 0, HH - 1),  cy1 = iclamp(iy1, 0, HH - 1);
            int cx0 = iclamp(ix0, 0, WW2 - 1), cx1 = iclamp(ix1, 0, WW2 - 1);
            int i00 = cy0 * WW2 + cx0, i01 = cy0 * WW2 + cx1;
            int i10 = cy1 * WW2 + cx0, i11 = cy1 * WW2 + cx1;
            float m00 = (1.f - wy) * (1.f - wx) * vy0 * vx0;
            float m01 = (1.f - wy) * wx         * vy0 * vx1;
            float m10 = wy         * (1.f - wx) * vy1 * vx0;
            float m11 = wy         * wx         * vy1 * vx1;
#pragma unroll
            for (int i = 0; i < 8; ++i) {
                int c = i * 16 + cg;
                const float* xc = srcb + (size_t)c * HW;
                float v = m00 * xc[i00] + m01 * xc[i01] + m10 * xc[i10] + m11 * xc[i11];
                samp[p * SROW + c * 9 + k] = (_Float16)v;
            }
        }
    }

    // ---------- phase 2: K-chunked WMMA GEMM (M=16 pix, N=16 ch per wave) ----
    const int lane  = tid & 31;
    const int wave  = tid >> 5;        // 0..7 -> out-channel tile
    const int hi    = lane >> 4;       // K half-select per ISA 16-bit layout
    const int m     = lane & 15;       // M (A) / N (B,D) index
    const int otile = wave * 16;

    const unsigned           ldsbase = (unsigned)(size_t)(&wlds[0]);
    const unsigned long long gbase   = (unsigned long long)(size_t)wh;

    v8f acc = {};
    for (int kc = 0; kc < CK; kc += KCH) {
        __syncthreads();               // wlds free for reuse + (first iter) samp ready
        // async global -> LDS copy of weight chunk: 128 rows x 64 halves (16 KB)
#pragma unroll
        for (int i = 0; i < 4; ++i) {
            int idx = i * 256 + tid;                       // 0..1023 granules of 16B
            int o = idx >> 3;                              // row 0..127
            int s = idx & 7;                               // 16B granule in row
            unsigned lb = ldsbase + (unsigned)((o * WROW + s * 8) * 2);
            unsigned go = (unsigned)((o * CK + kc + s * 8) * 2);
            asm volatile("global_load_async_to_lds_b128 %0, %1, %2"
                         :: "v"(lb), "v"(go), "s"(gbase)
                         : "memory");
        }
        asm volatile("s_wait_asynccnt 0x0" ::: "memory");
        __syncthreads();
#pragma unroll
        for (int j = 0; j < 2; ++j) {
            int kk = kc + j * 32;
            v8h a0 = *(const v8h*)(&samp[m * SROW + kk + hi * 8]);
            v8h a1 = *(const v8h*)(&samp[m * SROW + kk + 16 + hi * 8]);
            v8h b0 = *(const v8h*)(&wlds[(otile + m) * WROW + j * 32 + hi * 8]);
            v8h b1 = *(const v8h*)(&wlds[(otile + m) * WROW + j * 32 + 16 + hi * 8]);
            v16h av = __builtin_shufflevector(a0, a1, 0,1,2,3,4,5,6,7,8,9,10,11,12,13,14,15);
            v16h bv = __builtin_shufflevector(b0, b1, 0,1,2,3,4,5,6,7,8,9,10,11,12,13,14,15);
            acc = __builtin_amdgcn_wmma_f32_16x16x32_f16(false, av, false, bv,
                                                         (short)0, acc, false, false);
        }
    }

    // ---------- epilogue: bias + ReLU (+ residual), per ISA C/D layout -------
    const int ch = otile + m;                       // N = lane%16
    float bb = bias[ch];
    size_t obase = ((size_t)b * COUT + ch) * HW + pix0;
#pragma unroll
    for (int r = 0; r < 8; ++r) {
        int pm = r + hi * 8;                        // M = r + 8*(lane>=16)
        float v = acc[r] + bb;
        v = fmaxf(v, 0.f);
        if (ADD_RES) v += res[obase + pm];
        dst[obase + pm] = v;
    }
}

// ---------------------------------------------------------------------------
extern "C" void kernel_launch(void* const* d_in, const int* in_sizes, int n_in,
                              void* d_out, int out_size, void* d_ws, size_t ws_size,
                              hipStream_t stream) {
    const float* x   = (const float*)d_in[0];
    const float* ow1 = (const float*)d_in[1];
    const float* ob1 = (const float*)d_in[2];
    const float* dw1 = (const float*)d_in[3];
    const float* db1 = (const float*)d_in[4];
    const float* ow2 = (const float*)d_in[5];
    const float* ob2 = (const float*)d_in[6];
    const float* dw2 = (const float*)d_in[7];
    const float* db2 = (const float*)d_in[8];
    float* out = (float*)d_out;

    char* ws = (char*)d_ws;
    const size_t off_bytes = (size_t)NB * 18 * HW * sizeof(float);      // 1,327,104
    const size_t h1_bytes  = (size_t)NB * COUT * HW * sizeof(float);    // 9,437,184
    const size_t wh_bytes  = (size_t)COUT * CK * sizeof(_Float16);      //   294,912
    float*    offb = (float*)(ws);
    float*    h1   = (float*)(ws + off_bytes);
    _Float16* wh1  = (_Float16*)(ws + off_bytes + h1_bytes);
    _Float16* wh2  = (_Float16*)(ws + off_bytes + h1_bytes + wh_bytes);

    const int nblk_deform = NB * (HW / TILE_P);   // 1152
    const int nblk_off    = (NB * HW) / 256;      // 72
    const int nblk_wcast  = (2 * COUT * CK) / 256;// 1152

    wcast_kernel<<<nblk_wcast, 256, 0, stream>>>(dw1, dw2, wh1, wh2);
    offconv_kernel<<<nblk_off, 256, 0, stream>>>(x, ow1, ob1, offb);
    deform_kernel<false><<<nblk_deform, 256, 0, stream>>>(x, offb, wh1, db1, nullptr, h1);
    offconv_kernel<<<nblk_off, 256, 0, stream>>>(h1, ow2, ob2, offb);
    deform_kernel<true><<<nblk_deform, 256, 0, stream>>>(h1, offb, wh2, db2, x, out);
}
